// GraphFuse_36249523978496
// MI455X (gfx1250) — matrix operand order repaired and checked by reading, not compiled, not run on hardware
//
#include <hip/hip_runtime.h>

// ---------------------------------------------------------------------------
// MI455X (gfx1250) implementation of the linear-attention transformer block.
//
// Key simplifications derived from the reference:
//   * einsum('nhd,hdd->nhd') uses only diag(kvs) -> a [512] column reduction.
//   * Frobenius norms are global scalars -> folded into the epilogue.
// Dominant cost: 6 GEMMs (~412 GFLOP) vs ~2.2 GB HBM traffic -> memory bound
// at 23.3 TB/s; bf16 WMMA (v_wmma_f32_16x16x32_bf16) keeps compute far off
// the critical path while f32 accumulation preserves accuracy.
// ---------------------------------------------------------------------------

typedef __attribute__((ext_vector_type(16))) __bf16        v16bf;
typedef __attribute__((ext_vector_type(8)))  float         v8f;
typedef __attribute__((ext_vector_type(4)))  unsigned int  u32x4;

union Frag { v16bf v; u32x4 q[2]; };

__device__ __forceinline__ unsigned short f2bf(float f) {
    unsigned int u = __float_as_uint(f);
    unsigned int r = (u + 0x7FFFu + ((u >> 16) & 1u)) >> 16;   // RNE
    return (unsigned short)r;
}

// ---------------------------------------------------------------------------
// Weight convert + transpose: Wt[n*K + k] = bf16(W[k*N + n])
// ---------------------------------------------------------------------------
__global__ __launch_bounds__(256) void cvt_w_t(const float* __restrict__ W,
                                               unsigned short* __restrict__ Wt,
                                               int K, int N) {
    int idx = blockIdx.x * 256 + threadIdx.x;
    if (idx >= K * N) return;
    int n = idx % N, k = idx / N;
    Wt[(size_t)n * K + k] = f2bf(W[idx]);
}

// ---------------------------------------------------------------------------
// Row LayerNorm (D=512), fp32 in -> bf16 out. One block (256 thr) per row.
// ---------------------------------------------------------------------------
__global__ __launch_bounds__(256) void layernorm_bf16(
    const float* __restrict__ X, const float* __restrict__ g,
    const float* __restrict__ b, unsigned short* __restrict__ Y) {
    int row = blockIdx.x, t = threadIdx.x;
    const float* x = X + (size_t)row * 512;
    float2 v = *(const float2*)(x + 2 * t);
    __shared__ float r1[256], r2[256];
    r1[t] = v.x + v.y;
    r2[t] = v.x * v.x + v.y * v.y;
    __syncthreads();
    for (int off = 128; off > 0; off >>= 1) {
        if (t < off) { r1[t] += r1[t + off]; r2[t] += r2[t + off]; }
        __syncthreads();
    }
    float mean = r1[0] * (1.0f / 512.0f);
    float var  = r2[0] * (1.0f / 512.0f) - mean * mean;
    float inv  = rsqrtf(var + 1e-5f);
    float2 gg = *(const float2*)(g + 2 * t);
    float2 bb = *(const float2*)(b + 2 * t);
    unsigned short y0 = f2bf((v.x - mean) * inv * gg.x + bb.x);
    unsigned short y1 = f2bf((v.y - mean) * inv * gg.y + bb.y);
    Y[(size_t)row * 512 + 2 * t]     = y0;
    Y[(size_t)row * 512 + 2 * t + 1] = y1;
}

// ---------------------------------------------------------------------------
// bf16 WMMA GEMM: Out[M,Nc] = act( A[M,K] @ Bt[Nc,K]^T + bias ) (+Res)
// Block tile 128x128x32; 8 waves, each wave a 32x64 patch (2x4 wmma tiles).
// ---------------------------------------------------------------------------
template <int DO_GELU, int OUT_BF16, int HAS_RES>
__global__ __launch_bounds__(256) void gemm_wmma_bf16(
    const __bf16* __restrict__ A, const __bf16* __restrict__ Bt,
    const float* __restrict__ bias, const float* __restrict__ Res,
    void* __restrict__ Out, int M, int Nc, int K) {
    __shared__ __align__(16) __bf16 As[128][40];   // 32 + 8 pad (80B rows)
    __shared__ __align__(16) __bf16 Bs[128][40];

    const int tid  = threadIdx.x;
    const int wave = tid >> 5, lane = tid & 31;
    const int wr = wave >> 1, wc = wave & 1;       // 4x2 wave grid
    const int lr = lane & 15, grp = lane >> 4;
    const int m0 = blockIdx.y * 128, n0 = blockIdx.x * 128;

    v8f acc[2][4] = {};

    const int ldrow = tid >> 1;
    const int ldch  = (tid & 1) << 4;              // 0 or 16 (bf16 elems)

    for (int kt = 0; kt < K; kt += 32) {
        // ---- stage 128x32 A tile and 128x32 Bt tile into LDS -------------
        const u32x4* ga =
            (const u32x4*)(A + (size_t)(m0 + ldrow) * K + kt + ldch);
        const u32x4* gb =
            (const u32x4*)(Bt + (size_t)(n0 + ldrow) * K + kt + ldch);
        u32x4 a0 = ga[0], a1 = ga[1];
        u32x4 b0 = gb[0], b1 = gb[1];
        if (kt + 32 < K) {                          // prefetch next k-tile
            __builtin_prefetch(A + (size_t)(m0 + ldrow) * K + kt + 32 + ldch, 0, 0);
            __builtin_prefetch(Bt + (size_t)(n0 + ldrow) * K + kt + 32 + ldch, 0, 0);
        }
        *(u32x4*)(&As[ldrow][ldch])     = a0;
        *(u32x4*)(&As[ldrow][ldch + 8]) = a1;
        *(u32x4*)(&Bs[ldrow][ldch])     = b0;
        *(u32x4*)(&Bs[ldrow][ldch + 8]) = b1;
        __syncthreads();

        // ---- fragments per ISA layout tables -----------------------------
        Frag af[2], bfr[4];
#pragma unroll
        for (int mi = 0; mi < 2; ++mi) {
            int mr = wr * 32 + mi * 16 + lr;
            af[mi].q[0] = *(const u32x4*)(&As[mr][grp * 8]);        // k: g*8..
            af[mi].q[1] = *(const u32x4*)(&As[mr][16 + grp * 8]);   // k: 16+g*8..
        }
#pragma unroll
        for (int ni = 0; ni < 4; ++ni) {
            int nc = wc * 64 + ni * 16 + lr;
            bfr[ni].q[0] = *(const u32x4*)(&Bs[nc][grp * 8]);       // k: g*8..
            bfr[ni].q[1] = *(const u32x4*)(&Bs[nc][16 + grp * 8]);
        }
#pragma unroll
        for (int mi = 0; mi < 2; ++mi)
#pragma unroll
            for (int ni = 0; ni < 4; ++ni)
                acc[mi][ni] = __builtin_amdgcn_wmma_f32_16x16x32_bf16(
                    false, af[mi].v, false, bfr[ni].v,
                    (short)0, acc[mi][ni], false, false);
        __syncthreads();
    }

    // ---- epilogue: D layout -> lane = column, vgpr r -> row grp*8+r ------
#pragma unroll
    for (int mi = 0; mi < 2; ++mi)
#pragma unroll
        for (int ni = 0; ni < 4; ++ni) {
            int ng = n0 + wc * 64 + ni * 16 + lr;
            float bv = bias[ng];
#pragma unroll
            for (int r = 0; r < 8; ++r) {
                int mg = m0 + wr * 32 + mi * 16 + grp * 8 + r;
                float val = acc[mi][ni][r] + bv;
                if (DO_GELU)
                    val = 0.5f * val * (1.0f + erff(val * 0.70710678118f));
                size_t oi = (size_t)mg * Nc + ng;
                if (HAS_RES) val += Res[oi];
                if (OUT_BF16) ((unsigned short*)Out)[oi] = f2bf(val);
                else          ((float*)Out)[oi] = val;
            }
        }
}

// ---------------------------------------------------------------------------
// Stage 1 column reduction: per 128-row slab compute colsum(ks), diag(k*v),
// and partial sum(q^2), sum(k^2).  partial layout: [blk][1040] floats.
// ---------------------------------------------------------------------------
__global__ __launch_bounds__(256) void col_reduce_partial(
    const float* __restrict__ qs, const float* __restrict__ ks,
    const float* __restrict__ vs, float* __restrict__ partial, int rowsPerBlk) {
    int blk = blockIdx.x, t = threadIdx.x;
    int r0 = blk * rowsPerBlk;
    int c0 = t, c1 = t + 256;
    float cs0 = 0, cs1 = 0, cv0 = 0, cv1 = 0, sq = 0, sk = 0;
    for (int r = 0; r < rowsPerBlk; ++r) {
        size_t base = (size_t)(r0 + r) * 512;
        float k0 = ks[base + c0], k1 = ks[base + c1];
        float v0 = vs[base + c0], v1 = vs[base + c1];
        float q0 = qs[base + c0], q1 = qs[base + c1];
        cs0 += k0; cs1 += k1;
        cv0 += k0 * v0; cv1 += k1 * v1;
        sq += q0 * q0 + q1 * q1;
        sk += k0 * k0 + k1 * k1;
    }
    float* P = partial + (size_t)blk * 1040;
    P[c0] = cs0; P[c1] = cs1;
    P[512 + c0] = cv0; P[512 + c1] = cv1;
    __shared__ float rq[256], rk[256];
    rq[t] = sq; rk[t] = sk;
    __syncthreads();
    for (int off = 128; off > 0; off >>= 1) {
        if (t < off) { rq[t] += rq[t + off]; rk[t] += rk[t + off]; }
        __syncthreads();
    }
    if (t == 0) { P[1024] = rq[0]; P[1025] = rk[0]; }
}

// Stage 2: fold partials (fixed order -> deterministic).
__global__ __launch_bounds__(256) void col_reduce_final(
    const float* __restrict__ partial, float* __restrict__ stats, int nblk) {
    int t = threadIdx.x, c0 = t, c1 = t + 256;
    float a0 = 0, a1 = 0, b0 = 0, b1 = 0;
    for (int b = 0; b < nblk; ++b) {
        const float* P = partial + (size_t)b * 1040;
        a0 += P[c0]; a1 += P[c1];
        b0 += P[512 + c0]; b1 += P[512 + c1];
    }
    stats[c0] = a0; stats[c1] = a1;
    stats[512 + c0] = b0; stats[512 + c1] = b1;
    if (t == 0) { float s = 0; for (int b = 0; b < nblk; ++b) s += partial[(size_t)b * 1040 + 1024]; stats[1024] = s; }
    if (t == 1) { float s = 0; for (int b = 0; b < nblk; ++b) s += partial[(size_t)b * 1040 + 1025]; stats[1025] = s; }
}

// ---------------------------------------------------------------------------
// Attention epilogue per row:
//   num   = q*kv_diag*inv_nm + v*Nf
//   den_h = (q . colsum_ks)_head * inv_nm + Nf      (per 64-wide head)
//   attn  = bf16(num / den)
// ---------------------------------------------------------------------------
__global__ __launch_bounds__(256) void attn_epilogue(
    const float* __restrict__ qs, const float* __restrict__ vs,
    const float* __restrict__ stats, unsigned short* __restrict__ attn,
    float Nf) {
    int row = blockIdx.x, t = threadIdx.x;
    size_t base = (size_t)row * 512;
    float inv_nm = rsqrtf(stats[1024] * stats[1025]);   // 1/(|q|*|k|)
    __shared__ float prod[512];
    __shared__ float nrm[8];
    float q0 = qs[base + t], q1 = qs[base + t + 256];
    prod[t]       = q0 * stats[t];
    prod[t + 256] = q1 * stats[t + 256];
    __syncthreads();
    if (t < 8) {
        float s = 0;
#pragma unroll
        for (int i = 0; i < 64; ++i) s += prod[t * 64 + i];
        nrm[t] = s * inv_nm + Nf;
    }
    __syncthreads();
    float v0 = vs[base + t], v1 = vs[base + t + 256];
    float n0 = q0 * stats[512 + t] * inv_nm + v0 * Nf;
    float n1 = q1 * stats[512 + t + 256] * inv_nm + v1 * Nf;
    attn[base + t]       = f2bf(n0 / nrm[t >> 6]);
    attn[base + t + 256] = f2bf(n1 / nrm[(t + 256) >> 6]);
}

// ---------------------------------------------------------------------------
// Orchestration.  Workspace layout (liveness-based reuse, ~528 MB peak):
//   [0x000000, 0x600000)  bf16 W^T: Wq,Wk,Wv,Wh (512K ea), W1 (2M), W2 (2M)
//   [0x600000, 0x602000)  stats (colsum_ks | kv_diag | ssq_q | ssq_k)
//   [0x602000, 0x812000)  partial reductions [512][1040] f32
//   QS  f32 128MB @ 0x01000000   (later reused by G bf16 256MB)
//   KS  f32 128MB @ 0x09000000
//   VS  f32 128MB @ 0x11000000   (later reused by H_PRE f32 128MB)
//   LNQ bf16 64MB @ 0x19000000   (later reused by ATTN bf16)
//   LNS bf16 64MB @ 0x1D000000   (later reused by Y bf16)
// ---------------------------------------------------------------------------
extern "C" void kernel_launch(void* const* d_in, const int* in_sizes, int n_in,
                              void* d_out, int out_size, void* d_ws,
                              size_t ws_size, hipStream_t stream) {
    const float* q_in  = (const float*)d_in[0];
    const float* s_in  = (const float*)d_in[1];
    const float* Wq = (const float*)d_in[2];  const float* bq = (const float*)d_in[3];
    const float* Wk = (const float*)d_in[4];  const float* bk = (const float*)d_in[5];
    const float* Wv = (const float*)d_in[6];  const float* bv = (const float*)d_in[7];
    const float* Wh = (const float*)d_in[8];  const float* bh = (const float*)d_in[9];
    const float* lnkv_g = (const float*)d_in[10]; const float* lnkv_b = (const float*)d_in[11];
    const float* lnq_g  = (const float*)d_in[12]; const float* lnq_b  = (const float*)d_in[13];
    const float* ln2_g  = (const float*)d_in[14]; const float* ln2_b  = (const float*)d_in[15];
    const float* W1 = (const float*)d_in[16]; const float* b1 = (const float*)d_in[17];
    const float* W2 = (const float*)d_in[18]; const float* b2 = (const float*)d_in[19];

    const int D = 512, D4 = 2048;
    const int N = in_sizes[0] / D;            // 65536 rows
    const float Nf = (float)N;

    char* ws = (char*)d_ws;
    unsigned short* WqT = (unsigned short*)(ws + 0x000000);
    unsigned short* WkT = (unsigned short*)(ws + 0x080000);
    unsigned short* WvT = (unsigned short*)(ws + 0x100000);
    unsigned short* WhT = (unsigned short*)(ws + 0x180000);
    unsigned short* W1T = (unsigned short*)(ws + 0x200000);
    unsigned short* W2T = (unsigned short*)(ws + 0x400000);
    float* stats   = (float*)(ws + 0x600000);
    float* partial = (float*)(ws + 0x602000);
    float* QS   = (float*)(ws + 0x01000000);
    float* KS   = (float*)(ws + 0x09000000);
    float* VS   = (float*)(ws + 0x11000000);
    float* HPRE = (float*)(ws + 0x11000000);            // reuses VS
    unsigned short* LNQ  = (unsigned short*)(ws + 0x19000000);
    unsigned short* LNS  = (unsigned short*)(ws + 0x1D000000);
    unsigned short* ATTN = (unsigned short*)(ws + 0x19000000);  // reuses LNQ
    unsigned short* Ybf  = (unsigned short*)(ws + 0x1D000000);  // reuses LNS
    unsigned short* Gbf  = (unsigned short*)(ws + 0x01000000);  // reuses QS+KS

    // ---- 0. weights -> bf16 transposed -----------------------------------
    cvt_w_t<<<(D * D + 255) / 256, 256, 0, stream>>>(Wq, WqT, D, D);
    cvt_w_t<<<(D * D + 255) / 256, 256, 0, stream>>>(Wk, WkT, D, D);
    cvt_w_t<<<(D * D + 255) / 256, 256, 0, stream>>>(Wv, WvT, D, D);
    cvt_w_t<<<(D * D + 255) / 256, 256, 0, stream>>>(Wh, WhT, D, D);
    cvt_w_t<<<(D * D4 + 255) / 256, 256, 0, stream>>>(W1, W1T, D, D4);
    cvt_w_t<<<(D4 * D + 255) / 256, 256, 0, stream>>>(W2, W2T, D4, D);

    // ---- 1. layernorms ---------------------------------------------------
    layernorm_bf16<<<N, 256, 0, stream>>>(s_in, lnkv_g, lnkv_b, LNS);
    layernorm_bf16<<<N, 256, 0, stream>>>(q_in, lnq_g, lnq_b, LNQ);

    // ---- 2. QKV projections (bf16 WMMA, f32 out) -------------------------
    dim3 gD(D / 128, N / 128);
    gemm_wmma_bf16<0, 0, 0><<<gD, 256, 0, stream>>>(
        (const __bf16*)LNQ, (const __bf16*)WqT, bq, nullptr, QS, N, D, D);
    gemm_wmma_bf16<0, 0, 0><<<gD, 256, 0, stream>>>(
        (const __bf16*)LNS, (const __bf16*)WkT, bk, nullptr, KS, N, D, D);
    gemm_wmma_bf16<0, 0, 0><<<gD, 256, 0, stream>>>(
        (const __bf16*)LNS, (const __bf16*)WvT, bv, nullptr, VS, N, D, D);

    // ---- 3. global reductions (colsum_ks, diag(k*v), |q|^2, |k|^2) ------
    const int NBLK = 512, RPB = N / NBLK;
    col_reduce_partial<<<NBLK, 256, 0, stream>>>(QS, KS, VS, partial, RPB);
    col_reduce_final<<<1, 256, 0, stream>>>(partial, stats, NBLK);

    // ---- 4. attention epilogue -> bf16 -----------------------------------
    attn_epilogue<<<N, 256, 0, stream>>>(QS, VS, stats, ATTN, Nf);

    // ---- 5. h_pre = src + attn @ Wh + bh ---------------------------------
    gemm_wmma_bf16<0, 0, 1><<<gD, 256, 0, stream>>>(
        (const __bf16*)ATTN, (const __bf16*)WhT, bh, s_in, HPRE, N, D, D);

    // ---- 6. y = LN2(h_pre) -> bf16 ---------------------------------------
    layernorm_bf16<<<N, 256, 0, stream>>>(HPRE, ln2_g, ln2_b, Ybf);

    // ---- 7. g = gelu(y @ W1 + b1) -> bf16 --------------------------------
    dim3 g4(D4 / 128, N / 128);
    gemm_wmma_bf16<1, 1, 0><<<g4, 256, 0, stream>>>(
        (const __bf16*)Ybf, (const __bf16*)W1T, b1, nullptr, Gbf, N, D4, D);

    // ---- 8. out = h_pre + g @ W2 + b2 ------------------------------------
    gemm_wmma_bf16<0, 0, 1><<<gD, 256, 0, stream>>>(
        (const __bf16*)Gbf, (const __bf16*)W2T, b2, HPRE, (float*)d_out,
        N, D, D4);
}
